// Siamese_Searcher_43688407334918
// MI455X (gfx1250) — matrix operand order repaired and checked by reading, not compiled, not run on hardware
//
#include <hip/hip_runtime.h>
#include <hip/hip_bf16.h>

typedef __attribute__((ext_vector_type(16))) __bf16 v16bf;
typedef __attribute__((ext_vector_type(8)))  float  v8f;

#define GXc 64
#define GYc 32
#define GZc 4
#define Gc  8192          // GX*GY*GZ
#define Cc  128
#define Mc  64
#define NVOX 200000
#define DZc 10
#define DYc 400
#define DXc 352
#define NTAP 147          // 7*7*3
#define NTP  160          // padded to 10 WMMA n-tiles

// ---- grid-point geometry (matches reference _dense_grid_points / _rotate_z) ----
__device__ __forceinline__ void point_xyz(float cx, float cy, float cz,
                                          float dxe, float dye, float dze, float yaw,
                                          int p, float& X, float& Y, float& Z) {
    int iz = p & 3, iy = (p >> 2) & 31, ix = p >> 7;
    float lx = (ix + 0.5f) / 64.0f * dxe - 0.5f * dxe;
    float ly = (iy + 0.5f) / 32.0f * dye - 0.5f * dye;
    float lz = (iz + 0.5f) / 4.0f  * dze - 0.5f * dze;
    float s = sinf(yaw), c = cosf(yaw);
    X = lx * c - ly * s + cx;
    Y = lx * s + ly * c + cy;
    Z = lz + cz;
}

// ---- K1: v2p = -1 ----
__global__ void k_init_v2p(int* v2p, int n) {
    int i = blockIdx.x * blockDim.x + threadIdx.x;
    if (i < n) v2p[i] = -1;
}

// ---- K2: scatter voxel ids ----
__global__ void k_scatter(const int* __restrict__ coords, int* __restrict__ v2p) {
    int i = blockIdx.x * blockDim.x + threadIdx.x;
    if (i >= NVOX) return;
    int z = coords[i * 4 + 1], y = coords[i * 4 + 2], x = coords[i * 4 + 3];
    v2p[(z * DYc + y) * DXc + x] = i;
}

// ---- K3: W (1,7,7,3,128) f32 -> Wt[160][128] bf16 (taps >=147 zero) ----
__global__ void k_build_w(const float* __restrict__ proto, __bf16* __restrict__ Wt) {
    int i = blockIdx.x * blockDim.x + threadIdx.x;
    if (i >= NTP * Cc) return;
    int o = i >> 7, c = i & 127;
    float v = (o < NTAP) ? proto[o * Cc + c] : 0.0f;   // tap o=(kx*7+ky)*3+kz == flat layout
    Wt[i] = (__bf16)v;
}

// ---- K4: fused gather + WMMA GEMM:  D[m][q][o] = sum_c Vol[m][q][c] * W[c][o] ----
// block = 256 thr = 8 waves; block covers one ROI-tile of 128 grid points; wave -> 16 rows.
// D stored bf16 to halve the dominant HBM stream (write 160MB + re-read ~150MB).
__global__ __launch_bounds__(256) void k_stageA(const float* __restrict__ rois,
                                                const float* __restrict__ vfeats,
                                                const int*   __restrict__ v2p,
                                                const __bf16* __restrict__ Wt,
                                                __bf16* __restrict__ D) {
    int m    = blockIdx.x >> 6;
    int mt   = blockIdx.x & 63;
    int wave = threadIdx.x >> 5;
    int lane = threadIdx.x & 31;
    int row  = lane & 15;              // lanes l and l+16 share matrix row `row`
    int q    = mt * 128 + wave * 16 + row;

    // per-lane geometry + sparse lookup (redundant across the two lane halves)
    const float* roi = rois + m * 7;
    float wx, wy, wz;
    point_xyz(roi[0], roi[1], roi[2], roi[3] * 2.0f, roi[4] * 2.0f, roi[5], roi[6],
              q, wx, wy, wz);
    float fx = floorf((wx - 0.0f)  / 0.05f);
    float fy = floorf((wy + 40.0f) / 0.05f);
    float fz = floorf((wz + 3.0f)  / 0.1f);
    int vx = (int)floorf(fx / 4.0f);
    int vy = (int)floorf(fy / 4.0f);
    int vz = (int)floorf(fz / 4.0f);
    bool inb = (vz >= 0) & (vz < DZc) & (vy >= 0) & (vy < DYc) & (vx >= 0) & (vx < DXc);
    int cvx = min(max(vx, 0), DXc - 1);
    int cvy = min(max(vy, 0), DYc - 1);
    int cvz = min(max(vz, 0), DZc - 1);
    int pidx = v2p[(cvz * DYc + cvy) * DXc + cvx];
    bool valid = inb && (pidx >= 0);
    const float4* src = (const float4*)(vfeats + (size_t)max(pidx, 0) * Cc);

    // A fragments (16-bit A 16x32 layout): lane<16 holds K{0..7,16..23}, lane>=16 K{8..15,24..31}
    int half = (lane >> 4) * 8;
    v16bf a[4];
#pragma unroll
    for (int k = 0; k < 4; ++k) {
        int c0 = k * 32 + half;
        float4 u0 = src[(c0) >> 2];
        float4 u1 = src[(c0 + 4) >> 2];
        float4 u2 = src[(c0 + 16) >> 2];
        float4 u3 = src[(c0 + 20) >> 2];
        if (!valid) { u0 = u1 = u2 = u3 = make_float4(0.f, 0.f, 0.f, 0.f); }
        a[k][0]  = (__bf16)u0.x; a[k][1]  = (__bf16)u0.y; a[k][2]  = (__bf16)u0.z; a[k][3]  = (__bf16)u0.w;
        a[k][4]  = (__bf16)u1.x; a[k][5]  = (__bf16)u1.y; a[k][6]  = (__bf16)u1.z; a[k][7]  = (__bf16)u1.w;
        a[k][8]  = (__bf16)u2.x; a[k][9]  = (__bf16)u2.y; a[k][10] = (__bf16)u2.z; a[k][11] = (__bf16)u2.w;
        a[k][12] = (__bf16)u3.x; a[k][13] = (__bf16)u3.y; a[k][14] = (__bf16)u3.z; a[k][15] = (__bf16)u3.w;
    }

    // B layout (32x16, 16-bit): lane<16 -> col=row, K{0..7,16..23}; lane>=16 -> col=row, K{8..15,24..31}
    int koff = (lane >> 4) * 8;
    size_t drow0 = ((size_t)m * Gc + (size_t)mt * 128 + wave * 16 + (lane >> 4) * 8) * NTP;

    for (int nt = 0; nt < 10; ++nt) {
        const __bf16* wcol = Wt + (nt * 16 + row) * Cc + koff;
        v8f acc = {0.f, 0.f, 0.f, 0.f, 0.f, 0.f, 0.f, 0.f};
#pragma unroll
        for (int k = 0; k < 4; ++k) {
            typedef __attribute__((ext_vector_type(8))) __bf16 v8bf_t;
            v8bf_t blo = *(const v8bf_t*)(wcol + k * 32);
            v8bf_t bhi = *(const v8bf_t*)(wcol + k * 32 + 16);
            v16bf b = __builtin_shufflevector(blo, bhi, 0, 1, 2, 3, 4, 5, 6, 7, 8, 9, 10, 11, 12, 13, 14, 15);
            acc = __builtin_amdgcn_wmma_f32_16x16x32_bf16(
                false, a[k], false, b, (short)0, acc, false, false);
        }
        // C/D layout: lane half -> rows r / r+8, col = lane&15
        __bf16* dp = D + drow0 + nt * 16 + row;
#pragma unroll
        for (int r = 0; r < 8; ++r) dp[(size_t)r * NTP] = (__bf16)acc[r];
    }
}

// ---- K5a: shift-sum over 147 taps -> mm[m][p]  (2048 blocks for BW saturation) ----
__global__ __launch_bounds__(256) void k_convsum(const __bf16* __restrict__ D,
                                                 float* __restrict__ mm) {
    int m = blockIdx.x >> 5;                      // 32 blocks per ROI
    int p = (blockIdx.x & 31) * 256 + threadIdx.x;
    const __bf16* Dm = D + (size_t)m * Gc * NTP;
    int iz = p & 3, iy = (p >> 2) & 31, ix = p >> 7;
    float acc = 0.0f;
    for (int kx = 0; kx < 7; ++kx) {
        int jx = ix + kx - 3;
        if ((unsigned)jx >= (unsigned)GXc) continue;
        for (int ky = 0; ky < 7; ++ky) {
            int jy = iy + ky - 3;
            if ((unsigned)jy >= (unsigned)GYc) continue;
            const __bf16* dr = Dm + (size_t)((jx * GYc + jy) * GZc) * NTP + (kx * 7 + ky) * 3;
#pragma unroll
            for (int kz = 0; kz < 3; ++kz) {
                int jz = iz + kz - 1;
                if ((unsigned)jz >= (unsigned)GZc) continue;
                acc += (float)dr[(size_t)jz * NTP + kz];
            }
        }
    }
    mm[(size_t)m * Gc + p] = acc;
}

// ---- K5b: per-ROI first-occurrence argmax + emit 7-float box ----
__global__ __launch_bounds__(256) void k_argmax(const float* __restrict__ rois,
                                                const float* __restrict__ proto_box,
                                                const float* __restrict__ mm,
                                                float* __restrict__ out) {
    int m = blockIdx.x;
    int tid = threadIdx.x;
    const float* mmm = mm + (size_t)m * Gc;
    float best = -3.4e38f;
    int bidx = 0x7fffffff;
    for (int p = tid; p < Gc; p += 256) {         // p increasing -> strict > keeps first max
        float v = mmm[p];
        if (v > best) { best = v; bidx = p; }
    }
    __shared__ float sv[256];
    __shared__ int   si[256];
    sv[tid] = best; si[tid] = bidx;
    __syncthreads();
    for (int off = 128; off > 0; off >>= 1) {
        if (tid < off) {
            float v2 = sv[tid + off]; int i2 = si[tid + off];
            if (v2 > sv[tid] || (v2 == sv[tid] && i2 < si[tid])) { sv[tid] = v2; si[tid] = i2; }
        }
        __syncthreads();
    }
    if (tid == 0) {
        const float* roi = rois + m * 7;
        float X, Y, Z;
        point_xyz(roi[0], roi[1], roi[2], roi[3] * 2.0f, roi[4] * 2.0f, roi[5], roi[6],
                  si[0], X, Y, Z);
        out[m * 7 + 0] = X;
        out[m * 7 + 1] = Y;
        out[m * 7 + 2] = Z;
#pragma unroll
        for (int j = 3; j < 7; ++j) out[m * 7 + j] = proto_box[j];
    }
}

static inline size_t align256(size_t x) { return (x + 255) & ~(size_t)255; }

extern "C" void kernel_launch(void* const* d_in, const int* in_sizes, int n_in,
                              void* d_out, int out_size, void* d_ws, size_t ws_size,
                              hipStream_t stream) {
    const float* rois      = (const float*)d_in[0];
    const float* proto_box = (const float*)d_in[1];
    const float* vfeats    = (const float*)d_in[2];
    const float* fproto    = (const float*)d_in[3];
    const int*   vcoords   = (const int*)d_in[4];
    float* out = (float*)d_out;

    // workspace: v2p (5.632 MB) | Wt (40 KB) | D bf16 (160 MB) | mm (2 MB) => ~168 MB
    char* ws = (char*)d_ws;
    const int nv2p = DZc * DYc * DXc;
    int* v2p = (int*)ws;
    size_t off = align256((size_t)nv2p * sizeof(int));
    __bf16* Wt = (__bf16*)(ws + off);
    off += align256((size_t)NTP * Cc * sizeof(__bf16));
    __bf16* D = (__bf16*)(ws + off);
    off += align256((size_t)Mc * Gc * NTP * sizeof(__bf16));
    float* mm = (float*)(ws + off);

    k_init_v2p<<<(nv2p + 255) / 256, 256, 0, stream>>>(v2p, nv2p);
    k_scatter<<<(NVOX + 255) / 256, 256, 0, stream>>>(vcoords, v2p);
    k_build_w<<<(NTP * Cc + 255) / 256, 256, 0, stream>>>(fproto, Wt);
    k_stageA<<<Mc * (Gc / 128), 256, 0, stream>>>(rois, vfeats, v2p, Wt, D);
    k_convsum<<<Mc * (Gc / 256), 256, 0, stream>>>(D, mm);
    k_argmax<<<Mc, 256, 0, stream>>>(rois, proto_box, mm, out);
}
